// SelfAttention_59227599012084
// MI455X (gfx1250) — matrix operand order repaired and checked by reading, not compiled, not run on hardware
//
#include <hip/hip_runtime.h>

typedef __attribute__((ext_vector_type(16))) _Float16 v16h;
typedef __attribute__((ext_vector_type(8)))  _Float16 v8h;
typedef __attribute__((ext_vector_type(4)))  _Float16 v4h;
typedef __attribute__((ext_vector_type(8)))  float    v8f;

#define B_      2
#define S_      2048
#define HID_    4096
#define HD_     128
#define NH_     32
#define NKV_    8
#define GROUPS_ (NH_ / NKV_)
#define EPS_    1e-6f

// ---------------------------------------------------------------------------
// Elementwise f32 -> f16 convert (vectorized x4). n must be a multiple of 4.
// ---------------------------------------------------------------------------
__global__ __launch_bounds__(256)
void cvt_f32_f16_kernel(const float* __restrict__ src, _Float16* __restrict__ dst,
                        size_t n4) {
  const size_t i = (size_t)blockIdx.x * 256 + threadIdx.x;
  if (i >= n4) return;
  const float4 v = ((const float4*)src)[i];
  v4h h;
  h[0] = (_Float16)v.x; h[1] = (_Float16)v.y;
  h[2] = (_Float16)v.z; h[3] = (_Float16)v.w;
  ((v4h*)dst)[i] = h;
}

// ---------------------------------------------------------------------------
// GEMM: out[row, col] = sum_k A[row,k] * W[col,k]    (A, W in f16, K = 4096)
// Tile: 16 rows x 128 cols per block (8 waves, 16x16 WMMA tile each).
// MODE 0: f16 out, transposed (b, head, d, s)             (V projection)
// MODE 1: RMS-norm + RoPE epilogue, f16 (b, head, s, d)   (Q/K projection)
// MODE 2: f32 out, row-major (row, 4096)                  (O projection)
// ---------------------------------------------------------------------------
template <int MODE>
__global__ __launch_bounds__(256)
void gemm16_kernel(const _Float16* __restrict__ A, const _Float16* __restrict__ W,
                   void* __restrict__ outp,
                   const float* __restrict__ cosp, const float* __restrict__ sinp,
                   const float* __restrict__ nw, int nheads) {
  __shared__ _Float16 As[16][64];
  __shared__ float    Ot[16][128];
  __shared__ float    sc[16];

  const int tid  = threadIdx.x;
  const int wave = tid >> 5;
  const int lane = tid & 31;
  const int m    = lane & 15;
  const int hi   = lane >> 4;
  const int base = 16 * hi;

  const int row0 = blockIdx.x * 16;   // token row in [0, B*S)
  const int head = blockIdx.y;        // head (0/1) or 128-col chunk (2)
  const int b    = row0 / S_;
  const int s0   = row0 - b * S_;

  const int wcol = head * 128 + wave * 16 + m;       // W row == output column
  const _Float16* wrow = &W[(size_t)wcol * HID_];

  const int lr = tid >> 4;            // cooperative A-load row 0..15
  const int lc = (tid & 15) * 4;      // cooperative A-load col (x4)

  v8f acc = {};

  for (int k0 = 0; k0 < HID_; k0 += 64) {
    __syncthreads();
    *(v4h*)&As[lr][lc] = *(const v4h*)&A[(size_t)(row0 + lr) * HID_ + k0 + lc];
    __syncthreads();

    // prefetch next W chunk (global_prefetch_b8)
    __builtin_prefetch((const void*)&wrow[k0 + 64 + 16 * hi], 0, 3);

#pragma unroll
    for (int cc = 0; cc < 64; cc += 32) {
      // A fragment: lane holds row m, K elems {cc+base+j, cc+base+8+j}
      const v8h a0 = *(const v8h*)&As[m][cc + base];
      const v8h a1 = *(const v8h*)&As[m][cc + base + 8];
      v16h af;
#pragma unroll
      for (int j = 0; j < 8; ++j) { af[j] = a0[j]; af[j + 8] = a1[j]; }

      // B fragment: 16 contiguous f16 along K
      const v16h bf = *(const v16h*)&wrow[k0 + cc + 16 * hi];

      acc = __builtin_amdgcn_wmma_f32_16x16x32_f16(false, af, false, bf,
                                                   (short)0, acc, false, false);
    }
  }

  __syncthreads();
#pragma unroll
  for (int r = 0; r < 8; ++r)
    Ot[r + 8 * hi][wave * 16 + m] = acc[r];
  __syncthreads();

  if (MODE == 1) {
    // per-(token,head) RMS scale: wave w reduces rows 2w, 2w+1
#pragma unroll
    for (int rr = 0; rr < 2; ++rr) {
      const int row = wave * 2 + rr;
      float ss = 0.f;
#pragma unroll
      for (int i = 0; i < 4; ++i) {
        const float t = Ot[row][lane + 32 * i];
        ss += t * t;
      }
#pragma unroll
      for (int msk = 16; msk >= 1; msk >>= 1) ss += __shfl_xor(ss, msk, 32);
      if (lane == 0) sc[row] = rsqrtf(ss * (1.0f / 128.0f) + EPS_);
    }
    __syncthreads();
  }

#pragma unroll
  for (int i = 0; i < 8; ++i) {
    const int idx = tid + i * 256;
    const int row = idx >> 7;
    const int col = idx & 127;
    float val = Ot[row][col];
    if (MODE == 1) {
      const int   pc  = (col + 64) & 127;
      const float va  = val * sc[row] * nw[col];
      const float vb  = Ot[row][pc] * sc[row] * nw[pc];
      const float rot = (col < 64) ? -vb : vb;
      const size_t ci = ((size_t)(b * S_ + s0 + row)) * HD_ + col;
      val = va * cosp[ci] + rot * sinp[ci];
    }
    if (MODE == 2) {
      ((float*)outp)[(size_t)(row0 + row) * 4096 + head * 128 + col] = val;
    } else if (MODE == 0) {  // transposed: (b, head, d, s)
      const size_t oi =
          (((size_t)b * nheads + head) * HD_ + col) * S_ + (s0 + row);
      ((_Float16*)outp)[oi] = (_Float16)val;
    } else {                 // (b, head, s, d)
      const size_t oi =
          (((size_t)b * nheads + head) * S_ + (s0 + row)) * HD_ + col;
      ((_Float16*)outp)[oi] = (_Float16)val;
    }
  }
}

// ---------------------------------------------------------------------------
// Flash attention, causal, GQA. One wave per (b, h, 16-query tile).
// q: (B,NH,S,HD) f16   k: (B,NKV,S,HD) f16   vT: (B,NKV,HD,S) f16
// o: (B,S,NH,HD) f16
// ---------------------------------------------------------------------------
__global__ __launch_bounds__(32)
void attn_kernel(const _Float16* __restrict__ q, const _Float16* __restrict__ k,
                 const _Float16* __restrict__ vT, _Float16* __restrict__ o) {
  __shared__ _Float16 Pl[16 * 32];

  const int lane = threadIdx.x;
  const int m    = lane & 15;
  const int hi   = lane >> 4;

  const int q0 = blockIdx.x * 16;
  const int h  = blockIdx.y;
  const int b  = blockIdx.z;
  const int hk = h / GROUPS_;

  const _Float16* qh = q  + ((size_t)(b * NH_  + h ) * S_) * HD_;
  const _Float16* kh = k  + ((size_t)(b * NKV_ + hk) * S_) * HD_;
  const _Float16* vh = vT + ((size_t)(b * NKV_ + hk) * HD_) * S_;

  // Q fragments over 4 K-chunks of 32 dims
  v16h qa[4];
#pragma unroll
  for (int c = 0; c < 4; ++c) {
    const _Float16* qr = &qh[(size_t)(q0 + m) * HD_ + 32 * c];
    const v8h a0 = *(const v8h*)&qr[16 * hi];
    const v8h a1 = *(const v8h*)&qr[16 * hi + 8];
#pragma unroll
    for (int j = 0; j < 8; ++j) { qa[c][j] = a0[j]; qa[c][j + 8] = a1[j]; }
  }

  v8f   Oacc[8] = {};
  float mrow[8], lrow[8];
#pragma unroll
  for (int r = 0; r < 8; ++r) { mrow[r] = -1e30f; lrow[r] = 0.f; }

  const float scale = 0.08838834764831845f;  // 128^-0.5

  for (int kb = 0; kb < q0 + 16; kb += 32) {
    // S = Q K^T : 16 queries x 32 keys (two 16-key column groups)
    v8f sacc[2] = {};
#pragma unroll
    for (int g = 0; g < 2; ++g) {
      const int key = kb + 16 * g + m;
#pragma unroll
      for (int c = 0; c < 4; ++c) {
        const v16h kf = *(const v16h*)&kh[(size_t)key * HD_ + 32 * c + 16 * hi];
        sacc[g] = __builtin_amdgcn_wmma_f32_16x16x32_f16(
            false, qa[c], false, kf, (short)0, sacc[g], false, false);
      }
    }

    // online softmax (a score row lives in one VGPR across a 16-lane half)
    float alpha[8];
#pragma unroll
    for (int r = 0; r < 8; ++r) {
      const int qrow = q0 + r + 8 * hi;
      float s0v = sacc[0][r] * scale;
      float s1v = sacc[1][r] * scale;
      if (kb + m      > qrow) s0v = -1e30f;
      if (kb + 16 + m > qrow) s1v = -1e30f;
      float mx = fmaxf(s0v, s1v);
#pragma unroll
      for (int msk = 8; msk >= 1; msk >>= 1)
        mx = fmaxf(mx, __shfl_xor(mx, msk, 32));
      const float mnew = fmaxf(mrow[r], mx);
      alpha[r] = __expf(mrow[r] - mnew);
      mrow[r]  = mnew;
      const float p0 = __expf(s0v - mnew);
      const float p1 = __expf(s1v - mnew);
      float rs = p0 + p1;
#pragma unroll
      for (int msk = 8; msk >= 1; msk >>= 1) rs += __shfl_xor(rs, msk, 32);
      lrow[r] = lrow[r] * alpha[r] + rs;
      Pl[(r + 8 * hi) * 32 + m]      = (_Float16)p0;
      Pl[(r + 8 * hi) * 32 + 16 + m] = (_Float16)p1;
    }
#pragma unroll
    for (int t = 0; t < 8; ++t)
#pragma unroll
      for (int r = 0; r < 8; ++r) Oacc[t][r] *= alpha[r];

    // relayout P: C/D layout -> A fragment via LDS
    __syncthreads();
    v16h pf;
#pragma unroll
    for (int j = 0; j < 8; ++j) {
      pf[j]     = Pl[m * 32 + 16 * hi + j];
      pf[j + 8] = Pl[m * 32 + 16 * hi + 8 + j];
    }
    __syncthreads();

    // O += P V : 8 dim-tiles of 16; V^T gives contiguous 32B B-fragments
#pragma unroll
    for (int t = 0; t < 8; ++t) {
      const int dcol = 16 * t + m;
      const v16h vf = *(const v16h*)&vh[(size_t)dcol * S_ + kb + 16 * hi];
      Oacc[t] = __builtin_amdgcn_wmma_f32_16x16x32_f16(
          false, pf, false, vf, (short)0, Oacc[t], false, false);
    }
  }

  float invl[8];
#pragma unroll
  for (int r = 0; r < 8; ++r) invl[r] = 1.0f / lrow[r];
#pragma unroll
  for (int t = 0; t < 8; ++t) {
#pragma unroll
    for (int r = 0; r < 8; ++r) {
      const int srow = q0 + r + 8 * hi;
      o[(((size_t)(b * S_ + srow)) * NH_ + h) * HD_ + 16 * t + m] =
          (_Float16)(Oacc[t][r] * invl[r]);
    }
  }
}

// ---------------------------------------------------------------------------
extern "C" void kernel_launch(void* const* d_in, const int* in_sizes, int n_in,
                              void* d_out, int out_size, void* d_ws,
                              size_t ws_size, hipStream_t stream) {
  (void)in_sizes; (void)n_in; (void)out_size; (void)ws_size;

  const float* x    = (const float*)d_in[0];
  const float* cosp = (const float*)d_in[1];
  const float* sinp = (const float*)d_in[2];
  const float* Wq   = (const float*)d_in[3];
  const float* Wk   = (const float*)d_in[4];
  const float* Wv   = (const float*)d_in[5];
  const float* Wo   = (const float*)d_in[6];
  const float* qnw  = (const float*)d_in[7];
  const float* knw  = (const float*)d_in[8];

  const size_t nx  = (size_t)B_ * S_ * HID_;        // 16.8M
  const size_t nwq = (size_t)NH_  * HD_ * HID_;     // 16.8M
  const size_t nwk = (size_t)NKV_ * HD_ * HID_;     // 4.2M
  const size_t nq  = (size_t)B_ * NH_  * S_ * HD_;  // 16.8M
  const size_t nkv = (size_t)B_ * NKV_ * S_ * HD_;  // 4.2M

  char* ws = (char*)d_ws;
  _Float16* xh   = (_Float16*)ws;  ws += nx  * 2;
  _Float16* Wqh  = (_Float16*)ws;  ws += nwq * 2;
  _Float16* Wkh  = (_Float16*)ws;  ws += nwk * 2;
  _Float16* Wvh  = (_Float16*)ws;  ws += nwk * 2;
  _Float16* Woh  = (_Float16*)ws;  ws += nwq * 2;
  _Float16* q_ws = (_Float16*)ws;  ws += nq  * 2;
  _Float16* k_ws = (_Float16*)ws;  ws += nkv * 2;
  _Float16* vT   = (_Float16*)ws;  ws += nkv * 2;
  _Float16* o_h  = (_Float16*)ws;

  // 1) convert activations + weights to f16 once
  cvt_f32_f16_kernel<<<(nx  / 4 + 255) / 256, 256, 0, stream>>>(x,  xh,  nx  / 4);
  cvt_f32_f16_kernel<<<(nwq / 4 + 255) / 256, 256, 0, stream>>>(Wq, Wqh, nwq / 4);
  cvt_f32_f16_kernel<<<(nwk / 4 + 255) / 256, 256, 0, stream>>>(Wk, Wkh, nwk / 4);
  cvt_f32_f16_kernel<<<(nwk / 4 + 255) / 256, 256, 0, stream>>>(Wv, Wvh, nwk / 4);
  cvt_f32_f16_kernel<<<(nwq / 4 + 255) / 256, 256, 0, stream>>>(Wo, Woh, nwq / 4);

  const int rt = (B_ * S_) / 16;  // 256 row tiles

  // 2) projections (+ fused RMS-norm + RoPE for Q/K; V written transposed)
  gemm16_kernel<1><<<dim3(rt, NH_),  dim3(256), 0, stream>>>(
      xh, Wqh, (void*)q_ws, cosp, sinp, qnw, NH_);
  gemm16_kernel<1><<<dim3(rt, NKV_), dim3(256), 0, stream>>>(
      xh, Wkh, (void*)k_ws, cosp, sinp, knw, NKV_);
  gemm16_kernel<0><<<dim3(rt, NKV_), dim3(256), 0, stream>>>(
      xh, Wvh, (void*)vT, nullptr, nullptr, nullptr, NKV_);

  // 3) causal flash attention
  attn_kernel<<<dim3(S_ / 16, NH_, B_), dim3(32), 0, stream>>>(
      q_ws, k_ws, vT, o_h);

  // 4) output projection -> f32
  gemm16_kernel<2><<<dim3(rt, HID_ / 128), dim3(256), 0, stream>>>(
      o_h, Woh, d_out, nullptr, nullptr, nullptr, 0);
}
